// StamImu_25872882991388
// MI455X (gfx1250) — matrix operand order repaired and checked by reading, not compiled, not run on hardware
//
#include <hip/hip_runtime.h>
#include <math.h>

// ---------------------------------------------------------------------------
// Types for CDNA5 WMMA (gfx1250, wave32)
// ---------------------------------------------------------------------------
typedef __bf16 bf16_t;
typedef __attribute__((ext_vector_type(16))) __bf16 bf16x16;
typedef __attribute__((ext_vector_type(8)))  float  v8f;

__device__ __forceinline__ v8f wmma_bf(bf16x16 a, bf16x16 b, v8f c) {
  // D = A(16x32) x B(32x16) + C(16x16), f32 accumulate
  return __builtin_amdgcn_wmma_f32_16x16x32_bf16(
      false, a, false, b, (short)0, c, false, false);
}

// ---- fragment loaders -------------------------------------------------------
// A matrix 16x32 (MxK): lanes 0-15 -> M=lane, K=0..15;
// lanes 16-31 -> M=lane-16, K=16..31. element(m,k) = p[m*ms + k*ks]
// With ks==1 each lane reads 16 contiguous elements.
__device__ __forceinline__ bf16x16 a_frag_f32(const float* p, int ms, int ks, int kmax) {
  int lane = threadIdx.x & 31;
  int m = lane & 15;
  int kb = (lane & 16) ? 16 : 0;
  bf16x16 a;
#pragma unroll
  for (int i = 0; i < 8; ++i) {
    int k0 = kb + i, k1 = kb + 8 + i;
    a[i]     = (bf16_t)((k0 < kmax) ? p[m * ms + k0 * ks] : 0.0f);
    a[i + 8] = (bf16_t)((k1 < kmax) ? p[m * ms + k1 * ks] : 0.0f);
  }
  return a;
}
__device__ __forceinline__ bf16x16 a_frag_bf(const bf16_t* p, int ms, int ks) {
  int lane = threadIdx.x & 31;
  int m = lane & 15;
  int kb = (lane & 16) ? 16 : 0;
  bf16x16 a;
#pragma unroll
  for (int i = 0; i < 8; ++i) {
    a[i]     = p[m * ms + (kb + i) * ks];
    a[i + 8] = p[m * ms + (kb + 8 + i) * ks];
  }
  return a;
}
// B matrix 32x16 (KxN): lanes 0-15 -> N=lane, K=0..15; lanes 16-31 -> N=lane-16,
// K=16..31. element(k,n) = p[k*ks + n*ns]. With ks==1 each lane reads 16
// contiguous elements -> ds_load_b128 x2 (bf16) / x4 (f32).
__device__ __forceinline__ bf16x16 b_frag_f32(const float* p, int ks, int ns, int kmax) {
  int lane = threadIdx.x & 31;
  int n = lane & 15;
  int kb = (lane & 16) ? 16 : 0;
  bf16x16 b;
#pragma unroll
  for (int i = 0; i < 16; ++i) {
    int k = kb + i;
    b[i] = (bf16_t)((k < kmax) ? p[k * ks + n * ns] : 0.0f);
  }
  return b;
}
__device__ __forceinline__ bf16x16 b_frag_bf(const bf16_t* p, int ks, int ns) {
  int lane = threadIdx.x & 31;
  int n = lane & 15;
  int kb = (lane & 16) ? 16 : 0;
  bf16x16 b;
#pragma unroll
  for (int i = 0; i < 16; ++i) b[i] = p[(kb + i) * ks + n * ns];
  return b;
}
// C/D 16x16 f32: VGPR r: lanes 0-15 -> (M=r, N=lane); lanes 16-31 -> (M=r+8)
__device__ __forceinline__ void c_store_f32(float* p, int ms, v8f c) {
  int lane = threadIdx.x & 31;
  int n = lane & 15;
  int rb = (lane & 16) ? 8 : 0;
#pragma unroll
  for (int r = 0; r < 8; ++r) p[(rb + r) * ms + n] = c[r];
}
// transposed-layout C access: element(m,n) = p[n*ns + m]; per lane the 8 m's
// are contiguous bf16 -> single ds_load/store_b128.
__device__ __forceinline__ v8f c_load_bf_t(const bf16_t* p, int ns) {
  int lane = threadIdx.x & 31;
  int n = lane & 15;
  int rb = (lane & 16) ? 8 : 0;
  v8f c;
#pragma unroll
  for (int r = 0; r < 8; ++r) c[r] = (float)p[n * ns + rb + r];
  return c;
}
__device__ __forceinline__ void c_store_bf_t(bf16_t* p, int ns, v8f c) {
  int lane = threadIdx.x & 31;
  int n = lane & 15;
  int rb = (lane & 16) ? 8 : 0;
#pragma unroll
  for (int r = 0; r < 8; ++r) p[n * ns + rb + r] = (bf16_t)c[r];
}
__device__ __forceinline__ v8f add_col_bias(v8f c, const float* bias) {
  float bb = bias[threadIdx.x & 15];
#pragma unroll
  for (int r = 0; r < 8; ++r) c[r] += bb;
  return c;
}

// ---- math helpers -----------------------------------------------------------
__device__ __forceinline__ float gelu_f(float x) {
  float t = tanhf(0.79788456f * x * (1.0f + 0.044715f * x * x));
  return 0.5f * x * (1.0f + t);
}
__device__ __forceinline__ float gelu_bwd_f(float x) {
  float t = tanhf(0.79788456f * x * (1.0f + 0.044715f * x * x));
  return 0.5f * x * ((1.0f - t * t) * (0.79788456f + 0.1070322243f * x * x)) +
         0.5f * (1.0f + t);
}
__device__ __forceinline__ float rowsum16(float x) {
#pragma unroll
  for (int o = 8; o > 0; o >>= 1) x += __shfl_xor(x, o, 16);
  return x;
}

// ---------------------------------------------------------------------------
// Kernel 1: tiled bf16-WMMA GEMM  C[M,N] = op(A[M,K] @ W[K,N])
// block tile 128x64, BK=32, 256 threads = 8 waves (4x2), wave tile 32x32.
// Weight tile staged TRANSPOSED in LDS so B-fragment loads are contiguous.
// ---------------------------------------------------------------------------
__global__ __launch_bounds__(256) void gemm_bf16_kernel(
    const float* __restrict__ A, const float* __restrict__ W,
    float* __restrict__ C, int M, int N, int K, int fuse_gelu) {
  __shared__ bf16_t As[128 * 40];  // [m][k], stride 40
  __shared__ bf16_t Bs[64 * 40];   // [n][k] transposed, stride 40

  int bm = blockIdx.x, bn = blockIdx.y;
  int tid = threadIdx.x, wid = tid >> 5;
  int mw = wid >> 1, nw = wid & 1;

  v8f acc[2][2] = {};

  for (int k0 = 0; k0 < K; k0 += 32) {
    {  // stage A tile
      int r = tid >> 1, c0 = (tid & 1) * 16;
      const float* src = A + (size_t)(bm * 128 + r) * K + k0 + c0;
#pragma unroll
      for (int i = 0; i < 16; ++i) As[r * 40 + c0 + i] = (bf16_t)src[i];
      if (k0 + 32 < K) __builtin_prefetch(src + 32, 0, 1);
    }
    {  // stage W tile, transposed: Bs[n][k]
      int r = tid >> 3, c0 = (tid & 7) * 8;  // r = k row, c0 = n base
      const float* src = W + (size_t)(k0 + r) * N + bn * 64 + c0;
#pragma unroll
      for (int i = 0; i < 8; ++i) Bs[(c0 + i) * 40 + r] = (bf16_t)src[i];
      if (k0 + 32 < K) __builtin_prefetch(src + (size_t)32 * N, 0, 1);
    }
    __syncthreads();

    bf16x16 af0 = a_frag_bf(As + (mw * 32) * 40, 40, 1);
    bf16x16 af1 = a_frag_bf(As + (mw * 32 + 16) * 40, 40, 1);
    bf16x16 bf0 = b_frag_bf(Bs + (nw * 32) * 40, 1, 40);
    bf16x16 bf1 = b_frag_bf(Bs + (nw * 32 + 16) * 40, 1, 40);
    acc[0][0] = wmma_bf(af0, bf0, acc[0][0]);
    acc[0][1] = wmma_bf(af0, bf1, acc[0][1]);
    acc[1][0] = wmma_bf(af1, bf0, acc[1][0]);
    acc[1][1] = wmma_bf(af1, bf1, acc[1][1]);
    __syncthreads();
  }

  int lane = tid & 31, n = lane & 15, rb = (lane & 16) ? 8 : 0;
#pragma unroll
  for (int im = 0; im < 2; ++im)
#pragma unroll
    for (int in = 0; in < 2; ++in) {
      int row0 = bm * 128 + mw * 32 + im * 16 + rb;
      int col = bn * 64 + nw * 32 + in * 16 + n;
#pragma unroll
      for (int r = 0; r < 8; ++r) {
        float v = acc[im][in][r];
        if (fuse_gelu) v = gelu_f(v);
        C[(size_t)(row0 + r) * N + col] = v;
      }
    }
}

// ---------------------------------------------------------------------------
// Kernel 2: lr vector  lrv[row,h] = sigmoid(dot(hs[row,:], lrw[h,:]) + lrb[h])
// ---------------------------------------------------------------------------
__global__ __launch_bounds__(192) void lr_kernel(
    const float* __restrict__ hs, const float* __restrict__ lrw,
    const float* __restrict__ lrb, float* __restrict__ lrv) {
  __shared__ float xs[16 * 768];
  int row0 = blockIdx.x * 16;
  for (int i = threadIdx.x; i < 16 * 768; i += 192)
    xs[i] = hs[(size_t)row0 * 768 + i];
  __syncthreads();
  int rl = threadIdx.x / 12, h = threadIdx.x % 12;
  float s = lrb[h];
  const float* w = lrw + h * 768;
  const float* x = xs + rl * 768;
  for (int k = 0; k < 768; ++k) s += x[k] * w[k];
  lrv[(size_t)(row0 + rl) * 12 + h] = 1.0f / (1.0f + expf(-s));
}

// ---------------------------------------------------------------------------
// Kernel 3: TTT scan. One block per (b,h); fast weights persist in LDS across
// all 256 steps (~190KB -> needs CDNA5 320KB WGP LDS). 256 threads = 8 waves.
// Weights stored transposed (W1t[n][d] stride 68, W2t[c][r] stride 260) so
// every weight B-fragment except phase F's W2^T and every update C-fragment is lane-contiguous.
// K=16 operands kept in transposed stride-20 copies for contiguous frags.
// ---------------------------------------------------------------------------
__global__ __launch_bounds__(256) void scan_kernel(
    const float* __restrict__ XQg, const float* __restrict__ XKg,
    const float* __restrict__ XVg, const float* __restrict__ lrvg,
    const float* __restrict__ ltok, const float* __restrict__ nwg,
    const float* __restrict__ nbg, const float* __restrict__ W1g,
    const float* __restrict__ b1g, const float* __restrict__ W2g,
    const float* __restrict__ b2g, float* __restrict__ outg) {
  extern __shared__ char smem[];
  const int NH = 12, LL = 4096, WID = 768, HD = 64;

  int b = blockIdx.x / NH, h = blockIdx.x % NH;
  int tid = threadIdx.x, wid = tid >> 5;

  bf16_t* W1t = (bf16_t*)smem;        // [n=256][d=64] stride 68
  bf16_t* W2t = W1t + 256 * 68;       // [c=64][r=256] stride 260
  float* fb   = (float*)(W2t + 64 * 260);
  float* b1s  = fb;            fb += 256;
  float* b2s  = fb;            fb += 64;
  float* tok  = fb;            fb += 16;
  float* lrv  = fb;            fb += 16;
  float* evec = fb;            fb += 16;
  float* gam  = fb;            fb += 64;
  float* bet  = fb;            fb += 64;
  float* xq   = fb;            fb += 16 * 68;   // 16x64 stride 68
  float* xk   = fb;            fb += 16 * 68;
  float* xv   = fb;            fb += 16 * 68;
  float* z1   = fb;            fb += 16 * 260;  // Z1 -> Z1_bar -> X2_bar
  float* x2   = fb;            fb += 16 * 260;
  float* gz1  = fb;            fb += 16 * 260;  // raw gZ1 (pre-gelu_bwd)
  float* gz1t = fb;            fb += 256 * 20;  // gz1t[c][j] final gZ1^T
  float* sstt = fb;            fb += 256 * 20;  // (-e.*X)^T staging [c][j]
  float* z2   = fb;            fb += 16 * 68;
  float* gz2  = fb;            fb += 16 * 68;
  float* z2b  = fb;            fb += 16 * 68;
  float* gz2t = fb;            fb += 64 * 20;   // gz2t[d][j]
  float* attn = fb;            fb += 16 * 20;
  float* m1   = fb;            fb += 16 * 20;

  // ---- init persistent state ----
  for (int i = tid; i < 256 * 64; i += 256) {   // W1t[n][d] = W1[d][n]
    int nn = i >> 6, d = i & 63;
    W1t[nn * 68 + d] = (bf16_t)W1g[h * 16384 + d * 256 + nn];
  }
  for (int i = tid; i < 64 * 256; i += 256) {   // W2t[c][r] = W2[r][c]
    int cA = i >> 8, r = i & 255;
    W2t[cA * 260 + r] = (bf16_t)W2g[h * 16384 + r * 64 + cA];
  }
  b1s[tid] = b1g[h * 256 + tid];
  if (tid < 64) {
    b2s[tid] = b2g[h * 64 + tid];
    gam[tid] = nwg[h * 64 + tid];
    bet[tid] = nbg[h * 64 + tid];
  }
  if (tid < 16) tok[tid] = fmaxf(1.0f / (float)(tid + 1) + ltok[tid], 0.0f);
  __syncthreads();

  for (int n = 0; n < LL / 16; ++n) {
    size_t grow = (size_t)b * LL + (size_t)n * 16;

    // ---- A: load XQ/XK/XV mini tiles (16x64), lr vector ----
    {
      int i = tid >> 4, d0 = (tid & 15) * 4;
      size_t gi = (grow + i) * WID + h * HD + d0;
#pragma unroll
      for (int u = 0; u < 4; ++u) {
        xq[i * 68 + d0 + u] = XQg[gi + u];
        xk[i * 68 + d0 + u] = XKg[gi + u];
        xv[i * 68 + d0 + u] = XVg[gi + u];
      }
      if (n + 1 < LL / 16) {  // gfx1250 global_prefetch_b8 for next step
        __builtin_prefetch(XQg + gi + (size_t)16 * WID, 0, 1);
        __builtin_prefetch(XKg + gi + (size_t)16 * WID, 0, 1);
        __builtin_prefetch(XVg + gi + (size_t)16 * WID, 0, 1);
      }
    }
    if (tid < 16) {
      float lv = lrvg[(grow + tid) * NH + h];
      lrv[tid] = lv;
      evec[tid] = -tok[15] * lv * (1.0f / 64.0f);  // -last_eta
    }
    __syncthreads();

    // ---- B: Z1 = XK @ W1 + b1 : 8 waves x 2 N-tiles ----
    for (int t = 0; t < 2; ++t) {
      int nt = wid * 2 + t;
      v8f c = {};
      c = wmma_bf(a_frag_f32(xk, 68, 1, 32),      b_frag_bf(W1t + nt * 16 * 68, 1, 68), c);
      c = wmma_bf(a_frag_f32(xk + 32, 68, 1, 32), b_frag_bf(W1t + nt * 16 * 68 + 32, 1, 68), c);
      c = add_col_bias(c, b1s + nt * 16);
      c_store_f32(z1 + nt * 16, 260, c);
    }
    __syncthreads();

    // ---- C: X2 = gelu(Z1) ----
    for (int i = tid; i < 16 * 256; i += 256) {
      int r = i >> 8, cc = i & 255;
      x2[r * 260 + cc] = gelu_f(z1[r * 260 + cc]);
    }
    __syncthreads();

    // ---- D: Z2 = X2 @ W2 + b2 (waves 0-3) ; Attn1 raw (wave 4) ----
    if (wid < 4) {
      int nt = wid;
      v8f c = {};
#pragma unroll
      for (int ks = 0; ks < 8; ++ks)
        c = wmma_bf(a_frag_f32(x2 + ks * 32, 260, 1, 32),
                    b_frag_bf(W2t + nt * 16 * 260 + ks * 32, 1, 260), c);
      c = add_col_bias(c, b2s + nt * 16);
      c_store_f32(z2 + nt * 16, 68, c);
    } else if (wid == 4) {
      v8f c = {};
      c = wmma_bf(a_frag_f32(xq, 68, 1, 32),      b_frag_f32(xk, 1, 68, 32), c);
      c = wmma_bf(a_frag_f32(xq + 32, 68, 1, 32), b_frag_f32(xk + 32, 1, 68, 32), c);
      c_store_f32(attn, 20, c);
    }
    __syncthreads();
    {  // M1 = -eta .* (Attn1 + tril)
      int i = tid >> 4, j = tid & 15;
      float e = tok[i] * lrv[j] * (1.0f / 64.0f);
      m1[i * 20 + j] = (j <= i) ? -e * (attn[i * 20 + j] + 1.0f) : 0.0f;
    }
    __syncthreads();

    // ---- E: gZ2 = ln_l2_bwd(Z2, XV-XK) ; writes gz2 and gz2t ----
    {
      int i = tid >> 4, ls = tid & 15;
      float v[4], xh[4], g[4];
      float mu = 0.0f;
#pragma unroll
      for (int u = 0; u < 4; ++u) { v[u] = z2[i * 68 + ls * 4 + u]; mu += v[u]; }
      mu = rowsum16(mu) * (1.0f / 64.0f);
      float var = 0.0f;
#pragma unroll
      for (int u = 0; u < 4; ++u) { float d = v[u] - mu; var += d * d; }
      var = rowsum16(var) * (1.0f / 64.0f);
      float rstd = rsqrtf(var + 1e-6f);
      float s1 = 0.0f, s2 = 0.0f;
#pragma unroll
      for (int u = 0; u < 4; ++u) {
        int d = ls * 4 + u;
        xh[u] = (v[u] - mu) * rstd;
        float tgt = xv[i * 68 + d] - xk[i * 68 + d];
        g[u] = (gam[d] * xh[u] + bet[d] - tgt) * gam[d];
        s1 += g[u]; s2 += g[u] * xh[u];
      }
      s1 = rowsum16(s1); s2 = rowsum16(s2);
#pragma unroll
      for (int u = 0; u < 4; ++u) {
        int d = ls * 4 + u;
        float gv = (64.0f * g[u] - s1 - xh[u] * s2) * (rstd * (1.0f / 64.0f));
        gz2[i * 68 + d] = gv;
        gz2t[d * 20 + i] = gv;
      }
    }
    __syncthreads();

    // ---- F: gZ1 = (gZ2 @ W2^T) .* gelu_bwd(Z1) -> transposed gz1t ----
    for (int t = 0; t < 2; ++t) {
      int nt = wid * 2 + t;
      v8f c = {};
      c = wmma_bf(a_frag_f32(gz2, 68, 1, 32),      b_frag_bf(W2t + nt * 16, 260, 1), c);
      c = wmma_bf(a_frag_f32(gz2 + 32, 68, 1, 32), b_frag_bf(W2t + nt * 16 + 32 * 260, 260, 1), c);
      c_store_f32(gz1 + nt * 16, 260, c);
    }
    __syncthreads();
    for (int i = tid; i < 16 * 256; i += 256) {
      int r = i >> 8, cc = i & 255;
      gz1t[cc * 20 + r] = gz1[r * 260 + cc] * gelu_bwd_f(z1[r * 260 + cc]);
    }
    __syncthreads();

    // ---- G: Z1_bar = XQ@W1 + M1@gZ1 + b1 (overwrites z1) ----
    for (int t = 0; t < 2; ++t) {
      int nt = wid * 2 + t;
      v8f c = {};
      c = wmma_bf(a_frag_f32(xq, 68, 1, 32),      b_frag_bf(W1t + nt * 16 * 68, 1, 68), c);
      c = wmma_bf(a_frag_f32(xq + 32, 68, 1, 32), b_frag_bf(W1t + nt * 16 * 68 + 32, 1, 68), c);
      c = wmma_bf(a_frag_f32(m1, 20, 1, 16),      b_frag_f32(gz1t + nt * 16 * 20, 1, 20, 16), c);
      c = add_col_bias(c, b1s + nt * 16);
      c_store_f32(z1 + nt * 16, 260, c);
    }
    __syncthreads();
    for (int i = tid; i < 16 * 256; i += 256) {  // X2_bar = gelu(Z1_bar)
      int r = i >> 8, cc = i & 255;
      z1[r * 260 + cc] = gelu_f(z1[r * 260 + cc]);
    }
    __syncthreads();

    // ---- H: Attn2 raw = X2_bar @ X2^T (wave 0) ----
    if (wid == 0) {
      v8f c = {};
#pragma unroll
      for (int ks = 0; ks < 8; ++ks)
        c = wmma_bf(a_frag_f32(z1 + ks * 32, 260, 1, 32),
                    b_frag_f32(x2 + ks * 32, 1, 260, 32), c);
      c_store_f32(attn, 20, c);
    }
    __syncthreads();
    {  // M2 = -eta .* (Attn2 + tril)
      int i = tid >> 4, j = tid & 15;
      float e = tok[i] * lrv[j] * (1.0f / 64.0f);
      m1[i * 20 + j] = (j <= i) ? -e * (attn[i * 20 + j] + 1.0f) : 0.0f;
    }
    __syncthreads();

    // ---- I: Z2_bar (waves 0-3) ; stage (-e.*X1)^T (waves 4-7) ----
    if (wid < 4) {
      int nt = wid;
      v8f c = {};
#pragma unroll
      for (int ks = 0; ks < 8; ++ks)
        c = wmma_bf(a_frag_f32(z1 + ks * 32, 260, 1, 32),
                    b_frag_bf(W2t + nt * 16 * 260 + ks * 32, 1, 260), c);
      c = wmma_bf(a_frag_f32(m1, 20, 1, 16), b_frag_f32(gz2t + nt * 16 * 20, 1, 20, 16), c);
      c = add_col_bias(c, b2s + nt * 16);
      c_store_f32(z2b + nt * 16, 68, c);
    } else {
      for (int i = tid - 128; i < 16 * 64; i += 128) {
        int j = i >> 6, d = i & 63;
        sstt[d * 20 + j] = evec[j] * xk[j * 68 + d];
      }
    }
    __syncthreads();

    // ---- J: updates. W1 += (-e.*X1)^T @ gZ1 : 64 frags over 8 waves ----
    {
      int mt = wid >> 1, ntb = (wid & 1) * 8;
      for (int t = 0; t < 8; ++t) {
        int nt = ntb + t;
        bf16_t* wp = W1t + (nt * 16) * 68 + mt * 16;
        v8f c = c_load_bf_t(wp, 68);
        c = wmma_bf(a_frag_f32(sstt + mt * 16 * 20, 20, 1, 16),
                    b_frag_f32(gz1t + nt * 16 * 20, 1, 20, 16), c);
        c_store_bf_t(wp, 68, c);
      }
    }
    __syncthreads();
    {  // b1 += sum_j (-e_j) * gZ1[j,:]
      float s = 0.0f;
#pragma unroll
      for (int j = 0; j < 16; ++j) s += evec[j] * gz1t[tid * 20 + j];
      b1s[tid] += s;
    }
    __syncthreads();
    for (int i = tid; i < 16 * 256; i += 256) {  // restage (-e.*X2)^T
      int j = i >> 8, cc = i & 255;
      sstt[cc * 20 + j] = evec[j] * x2[j * 260 + cc];
    }
    __syncthreads();
    {  // W2 += (-e.*X2)^T @ gZ2 : 64 frags over 8 waves
      for (int mt = wid * 2; mt < wid * 2 + 2; ++mt)
        for (int nt = 0; nt < 4; ++nt) {
          bf16_t* wp = W2t + (nt * 16) * 260 + mt * 16;
          v8f c = c_load_bf_t(wp, 260);
          c = wmma_bf(a_frag_f32(sstt + mt * 16 * 20, 20, 1, 16),
                      b_frag_f32(gz2t + nt * 16 * 20, 1, 20, 16), c);
          c_store_bf_t(wp, 260, c);
        }
    }
    if (tid < 64) {  // b2 += sum_j (-e_j) * gZ2[j,:]
      float s = 0.0f;
#pragma unroll
      for (int j = 0; j < 16; ++j) s += evec[j] * gz2t[tid * 20 + j];
      b2s[tid] += s;
    }
    __syncthreads();

    // ---- K: out = XQ + LayerNorm(Z2_bar; gamma, beta) ----
    {
      int i = tid >> 4, ls = tid & 15;
      float v[4];
      float mu = 0.0f;
#pragma unroll
      for (int u = 0; u < 4; ++u) { v[u] = z2b[i * 68 + ls * 4 + u]; mu += v[u]; }
      mu = rowsum16(mu) * (1.0f / 64.0f);
      float var = 0.0f;
#pragma unroll
      for (int u = 0; u < 4; ++u) { float d = v[u] - mu; var += d * d; }
      var = rowsum16(var) * (1.0f / 64.0f);
      float rstd = rsqrtf(var + 1e-6f);
#pragma unroll
      for (int u = 0; u < 4; ++u) {
        int d = ls * 4 + u;
        float o = xq[i * 68 + d] + gam[d] * ((v[u] - mu) * rstd) + bet[d];
        outg[(grow + i) * WID + h * HD + d] = o;
      }
    }
    __syncthreads();
  }
}

// ---------------------------------------------------------------------------
// Kernel 4: gate[r,:] *= LayerNorm(scan_out[r,:]; post_nw, post_nb)
// ---------------------------------------------------------------------------
__global__ __launch_bounds__(256) void lnmul_kernel(
    const float* __restrict__ x, float* __restrict__ gate,
    const float* __restrict__ w, const float* __restrict__ bb) {
  __shared__ float red[256];
  size_t row = blockIdx.x;
  const float* xr = x + row * 768;
  int tid = threadIdx.x;

  float s = 0.0f;
  for (int c = tid; c < 768; c += 256) s += xr[c];
  red[tid] = s; __syncthreads();
  for (int o = 128; o > 0; o >>= 1) { if (tid < o) red[tid] += red[tid + o]; __syncthreads(); }
  float mu = red[0] * (1.0f / 768.0f);
  __syncthreads();

  float s2 = 0.0f;
  for (int c = tid; c < 768; c += 256) { float d = xr[c] - mu; s2 += d * d; }
  red[tid] = s2; __syncthreads();
  for (int o = 128; o > 0; o >>= 1) { if (tid < o) red[tid] += red[tid + o]; __syncthreads(); }
  float rstd = rsqrtf(red[0] * (1.0f / 768.0f) + 1e-6f);
  __syncthreads();

  for (int c = tid; c < 768; c += 256) {
    float ln = w[c] * ((xr[c] - mu) * rstd) + bb[c];
    gate[row * 768 + c] *= ln;
  }
}

// ---------------------------------------------------------------------------
// Host launcher
// ---------------------------------------------------------------------------
extern "C" void kernel_launch(void* const* d_in, const int* in_sizes, int n_in,
                              void* d_out, int out_size, void* d_ws, size_t ws_size,
                              hipStream_t stream) {
  (void)in_sizes; (void)n_in; (void)out_size; (void)ws_size;
  const float* hs   = (const float*)d_in[0];
  const float* Wq   = (const float*)d_in[1];
  const float* Wk   = (const float*)d_in[2];
  const float* Wv   = (const float*)d_in[3];
  const float* Wo   = (const float*)d_in[4];
  const float* Wg   = (const float*)d_in[5];
  const float* lrw  = (const float*)d_in[6];
  const float* lrb  = (const float*)d_in[7];
  const float* ltok = (const float*)d_in[8];
  const float* nw   = (const float*)d_in[9];
  const float* nb   = (const float*)d_in[10];
  const float* pnw  = (const float*)d_in[11];
  const float* pnb  = (const float*)d_in[12];
  const float* W1   = (const float*)d_in[13];
  const float* b1   = (const float*)d_in[14];
  const float* W2   = (const float*)d_in[15];
  const float* b2   = (const float*)d_in[16];
  float* out = (float*)d_out;

  const int M = 8 * 4096, N = 768, K = 768;
  const size_t NE = (size_t)M * N;

  float* XQ   = (float*)d_ws;
  float* XK   = XQ + NE;
  float* XV   = XK + NE;
  float* GATE = XV + NE;
  float* SOUT = GATE + NE;
  float* LRV  = SOUT + NE;   // M * 12 floats

  dim3 ggrid(M / 128, N / 64);
  gemm_bf16_kernel<<<ggrid, 256, 0, stream>>>(hs, Wq, XQ, M, N, K, 0);
  gemm_bf16_kernel<<<ggrid, 256, 0, stream>>>(hs, Wk, XK, M, N, K, 0);
  gemm_bf16_kernel<<<ggrid, 256, 0, stream>>>(hs, Wv, XV, M, N, K, 0);
  gemm_bf16_kernel<<<ggrid, 256, 0, stream>>>(hs, Wg, GATE, M, N, K, 1);

  lr_kernel<<<M / 16, 192, 0, stream>>>(hs, lrw, lrb, LRV);

  // dynamic LDS: transposed bf16 weights (padded) + f32 scratch
  const size_t SCAN_LDS =
      (size_t)(256 * 68 + 64 * 260) * sizeof(bf16_t) +   // 68096 B
      (size_t)(256 + 64 + 16 + 16 + 16 + 64 + 64 +
               3 * 16 * 68 + 3 * 16 * 260 + 2 * 256 * 20 +
               3 * 16 * 68 + 64 * 20 + 2 * 16 * 20) * sizeof(float);
  scan_kernel<<<96, 256, SCAN_LDS, stream>>>(XQ, XK, XV, LRV, ltok, nw, nb,
                                             W1, b1, W2, b2, SOUT);

  lnmul_kernel<<<M, 256, 0, stream>>>(SOUT, GATE, pnw, pnb);

  gemm_bf16_kernel<<<ggrid, 256, 0, stream>>>(GATE, Wo, out, M, N, K, 0);
}
